// AttentionRNN_2697239462681
// MI455X (gfx1250) — matrix operand order repaired and checked by reading, not compiled, not run on hardware
//
#include <hip/hip_runtime.h>
#include <hip/hip_bf16.h>

// Problem constants
#define Bsz 64
#define Tsz 512
#define Esz 512
#define Hsz 1024
#define Csz 16

typedef __attribute__((ext_vector_type(16))) __bf16 v16bf;
typedef __attribute__((ext_vector_type(8)))  float  v8f;
typedef __attribute__((address_space(3))) unsigned short lds_us;

union AV {
    v16bf v;
    uint4 q[2];
};

__device__ __forceinline__ unsigned short f2bf(float f) {
    unsigned u = __float_as_uint(f);
    u += 0x7FFFu + ((u >> 16) & 1u);   // round-to-nearest-even
    return (unsigned short)(u >> 16);
}
__device__ __forceinline__ float bf2f(unsigned short s) {
    return __uint_as_float(((unsigned)s) << 16);
}

// ---------------------------------------------------------------------------
// init: zero h ping-pong buffers + grid-barrier state (deterministic per call)
// ---------------------------------------------------------------------------
__global__ __launch_bounds__(256) void k_init(unsigned* hbuf_u32, unsigned* bar) {
    unsigned i = blockIdx.x * 256u + threadIdx.x;
    if (i < (2u * Bsz * Hsz) / 2u)   // 2 buffers * 64*1024 ushort = 65536 uints
        hbuf_u32[i] = 0u;
    if (i < 2u) bar[i] = 0u;
}

// fp32 -> bf16 bulk convert
__global__ __launch_bounds__(256) void k_cvt(const float* __restrict__ src,
                                             unsigned short* __restrict__ dst, int n) {
    int i = blockIdx.x * 256 + threadIdx.x;
    if (i < n) dst[i] = f2bf(src[i]);
}

// ---------------------------------------------------------------------------
// Embedding gather: Xbf[m=t*B+b][e] = bf16(emb[inputs[b][t]][e])
// ---------------------------------------------------------------------------
__global__ __launch_bounds__(256) void k_gather(const int* __restrict__ inputs,
                                                const float* __restrict__ emb,
                                                unsigned short* __restrict__ Xbf) {
    unsigned idx = blockIdx.x * 256u + threadIdx.x;         // one per 4 elements
    if (idx >= (unsigned)(Bsz * Tsz * Esz / 4)) return;
    unsigned m = idx >> 7;                                  // E/4 = 128 chunks/row
    unsigned r = (idx & 127u) * 4u;
    unsigned t = m >> 6, b = m & 63u;
    int tok = inputs[b * Tsz + t];
    const float4 f = *(const float4*)(emb + (size_t)tok * Esz + r);
    uint2 p;
    p.x = (unsigned)f2bf(f.x) | ((unsigned)f2bf(f.y) << 16);
    p.y = (unsigned)f2bf(f.z) | ((unsigned)f2bf(f.w) << 16);
    *(uint2*)&Xbf[(size_t)m * Esz + r] = p;
}

// ---------------------------------------------------------------------------
// Batched input projection: Xp[m][n] = sum_k Xbf[m][k] * W_ih[n][k] + b_ih[n]
// M=32768, N=1024, K=512.  Grid (M/64, N/64), block 128 (4 waves).
// Each wave: 16(M) x 64(N) tile, 4 accumulators, K-loop step 32,
// register double-buffered so next-chunk loads overlap current WMMAs.
// ---------------------------------------------------------------------------
__global__ __launch_bounds__(128) void k_gemm_xp(const unsigned short* __restrict__ Xbf,
                                                 const unsigned short* __restrict__ Wih,
                                                 const float* __restrict__ b_ih,
                                                 float* __restrict__ Xp) {
    const int wave = threadIdx.x >> 5, lane = threadIdx.x & 31;
    const int laneLo = lane & 15, laneHi = lane >> 4;
    const int m0 = blockIdx.x * 64 + wave * 16;
    const int n0 = blockIdx.y * 64;

    const unsigned short* arow = Xbf + (size_t)(m0 + laneLo) * Esz;
    auto ldA = [&](AV& a, int k0) {
        a.q[0] = *(const uint4*)&arow[k0 + laneHi * 8];
        a.q[1] = *(const uint4*)&arow[k0 + 16 + laneHi * 8];
    };
    auto ldB = [&](AV* bm, int k0) {
#pragma unroll
        for (int j = 0; j < 4; ++j) {
            const unsigned short* brow = Wih + (size_t)(n0 + j * 16 + laneLo) * Esz;
            bm[j].q[0] = *(const uint4*)&brow[k0 + laneHi * 8];
            bm[j].q[1] = *(const uint4*)&brow[k0 + 16 + laneHi * 8];
        }
    };

    v8f acc[4] = {};
    AV aC, bC[4];
    ldA(aC, 0);
    ldB(bC, 0);
#pragma unroll
    for (int k0 = 0; k0 < Esz - 32; k0 += 32) {
        AV aN, bN[4];
        ldA(aN, k0 + 32);                       // in flight while we compute k0
        ldB(bN, k0 + 32);
#pragma unroll
        for (int j = 0; j < 4; ++j)
            acc[j] = __builtin_amdgcn_wmma_f32_16x16x32_bf16(
                false, aC.v, false, bC[j].v, (short)0, acc[j], false, false);
        aC = aN;
#pragma unroll
        for (int j = 0; j < 4; ++j) bC[j] = bN[j];
    }
#pragma unroll
    for (int j = 0; j < 4; ++j)
        acc[j] = __builtin_amdgcn_wmma_f32_16x16x32_bf16(
            false, aC.v, false, bC[j].v, (short)0, acc[j], false, false);

#pragma unroll
    for (int j = 0; j < 4; ++j) {
        int n = n0 + j * 16 + laneLo;
        float bias = b_ih[n];
#pragma unroll
        for (int v = 0; v < 8; ++v) {
            int m = m0 + laneHi * 8 + v;
            Xp[(size_t)m * Hsz + n] = acc[j][v] + bias;
        }
    }
}

// ---------------------------------------------------------------------------
// Persistent recurrent kernel.  64 WGs (one 16-col N-slice each) x 128 thr.
// W_hh slice loaded into LDS once via async global->LDS DMA (ASYNCcnt);
// h ping-pongs in global (L2-resident, 128 KB bf16).
// ---------------------------------------------------------------------------
__device__ __forceinline__ void grid_barrier(unsigned* cnt, volatile unsigned* gen,
                                             unsigned nwg) {
    __threadfence();                 // all threads: release stores to device scope
    __syncthreads();
    if (threadIdx.x == 0) {
        unsigned g = *gen;
        if (atomicAdd(cnt, 1u) == nwg - 1u) {
            *cnt = 0u;
            __threadfence();
            *gen = g + 1u;
        } else {
            while (*gen == g) { __builtin_amdgcn_s_sleep(2); }
        }
        __threadfence();             // acquire
    }
    __syncthreads();
}

__global__ __launch_bounds__(128) void k_rnn(const unsigned short* __restrict__ Whh,
                                             const float* __restrict__ Xp,
                                             const float* __restrict__ b_hh,
                                             const int* __restrict__ seq_lengths,
                                             unsigned short* hbuf,          // [2][B][H]
                                             unsigned short* __restrict__ obf, // [T][B][H]
                                             unsigned* bar_cnt, unsigned* bar_gen) {
    __shared__ unsigned short lB[16][Hsz + 8];   // +8 ushort pad vs bank conflicts
    const int n0 = blockIdx.x * 16;
    const int tid = threadIdx.x;
    const int wave = tid >> 5, lane = tid & 31;
    const int laneLo = lane & 15, laneHi = lane >> 4;
    const int m0 = wave * 16;                    // 4 waves cover B=64

    // Async DMA this WG's 16 rows of W_hh into LDS (32 KB), 16 B per issue.
    for (int idx = tid; idx < 16 * (Hsz / 8); idx += 128) {
        int r = idx >> 7;                        // H/8 = 128 chunks per row
        int c = (idx & 127) * 8;
        unsigned loff = (unsigned)(size_t)(lds_us*)&lB[r][c];
        unsigned long long ga =
            (unsigned long long)(size_t)&Whh[(size_t)(n0 + r) * Hsz + c];
        asm volatile("global_load_async_to_lds_b128 %0, %1, off"
                     :: "v"(loff), "v"(ga) : "memory");
    }
    asm volatile("s_wait_asynccnt 0x0" ::: "memory");
    __syncthreads();

    const int n = n0 + laneLo;
    const float bhh = b_hh[n];
    int lens[8];
#pragma unroll
    for (int v = 0; v < 8; ++v) lens[v] = seq_lengths[m0 + laneHi * 8 + v];

    for (int t = 0; t < Tsz; ++t) {
        const unsigned short* hsrc = hbuf + (size_t)(t & 1) * Bsz * Hsz;
        unsigned short*       hdst = hbuf + (size_t)((t + 1) & 1) * Bsz * Hsz;
        const unsigned short* arow = hsrc + (size_t)(m0 + laneLo) * Hsz;
        const unsigned short* brow = &lB[laneLo][0];

        // warm next step's Xp lines while the WMMA chain runs
        if (t + 1 < Tsz) {
            const float* xp_n = Xp + (size_t)(t + 1) * Bsz * Hsz;
#pragma unroll
            for (int v = 0; v < 8; ++v)
                __builtin_prefetch(&xp_n[(size_t)(m0 + laneHi * 8 + v) * Hsz + n], 0, 1);
        }

        v8f acc = {};
        AV aC;
        aC.q[0] = *(const uint4*)&arow[laneHi * 8];
        aC.q[1] = *(const uint4*)&arow[16 + laneHi * 8];
#pragma unroll
        for (int k0 = 0; k0 < Hsz; k0 += 32) {
            AV aN;
            if (k0 + 32 < Hsz) {                 // prefetch next A chunk (global)
                aN.q[0] = *(const uint4*)&arow[k0 + 32 + laneHi * 8];
                aN.q[1] = *(const uint4*)&arow[k0 + 48 + laneHi * 8];
            } else {
                aN = aC;
            }
            AV bm;                               // B chunk from LDS (fast)
            bm.q[0] = *(const uint4*)&brow[k0 + laneHi * 8];
            bm.q[1] = *(const uint4*)&brow[k0 + 16 + laneHi * 8];
            acc = __builtin_amdgcn_wmma_f32_16x16x32_bf16(
                false, aC.v, false, bm.v, (short)0, acc, false, false);
            aC = aN;
        }

        const float* xp_t = Xp + (size_t)t * Bsz * Hsz;
#pragma unroll
        for (int v = 0; v < 8; ++v) {
            int m = m0 + laneHi * 8 + v;
            float hn = tanhf(acc[v] + xp_t[(size_t)m * Hsz + n] + bhh);
            bool valid = t < lens[v];
            float hp = bf2f(hsrc[(size_t)m * Hsz + n]);
            hdst[(size_t)m * Hsz + n] = f2bf(valid ? hn : hp);
            obf[((size_t)t * Bsz + m) * Hsz + n] = f2bf(valid ? hn : 0.0f);
        }
        grid_barrier(bar_cnt, bar_gen, gridDim.x);
    }
    // final state ends in buffer 0 (T even)
}

// ---------------------------------------------------------------------------
// u[b][k] = sum_h state[b][h] * Wa[h][k]   (state = hbuf[0], bf16)
// ---------------------------------------------------------------------------
__global__ __launch_bounds__(256) void k_u(const unsigned short* __restrict__ state,
                                           const float* __restrict__ Wa,
                                           float* __restrict__ u) {
    int b = blockIdx.y;
    int k = blockIdx.x * 256 + threadIdx.x;
    float s = 0.f;
    for (int h = 0; h < Hsz; ++h)
        s += bf2f(state[(size_t)b * Hsz + h]) * Wa[(size_t)h * Hsz + k];
    u[(size_t)b * Hsz + k] = s;
}

// ---------------------------------------------------------------------------
// energy + softmax per batch row (ba term is constant per-row -> cancels)
// ---------------------------------------------------------------------------
__global__ __launch_bounds__(256) void k_att(const unsigned short* __restrict__ obf,
                                             const float* __restrict__ u,
                                             const int* __restrict__ inputs,
                                             const int* __restrict__ pad_tok,
                                             float* __restrict__ w) {
    __shared__ float us[Hsz];
    __shared__ float e[Tsz];
    __shared__ float red[256];
    const int b = blockIdx.x;
    for (int i = threadIdx.x; i < Hsz; i += 256) us[i] = u[(size_t)b * Hsz + i];
    __syncthreads();
    const int pad = *pad_tok;

    for (int t = threadIdx.x; t < Tsz; t += 256) {
        const unsigned short* o = obf + ((size_t)t * Bsz + b) * Hsz;
        float s = 0.f;
        for (int h8 = 0; h8 < Hsz / 8; ++h8) {
            uint4 q = *(const uint4*)&o[h8 * 8];
            const unsigned* qq = (const unsigned*)&q;
#pragma unroll
            for (int j = 0; j < 4; ++j) {
                s += bf2f((unsigned short)(qq[j] & 0xFFFFu)) * us[h8 * 8 + 2 * j];
                s += bf2f((unsigned short)(qq[j] >> 16))     * us[h8 * 8 + 2 * j + 1];
            }
        }
        e[t] = (inputs[b * Tsz + t] == pad) ? s - 1e6f : s;
    }
    __syncthreads();

    float lmax = -3.0e38f;
    for (int t = threadIdx.x; t < Tsz; t += 256) lmax = fmaxf(lmax, e[t]);
    red[threadIdx.x] = lmax; __syncthreads();
    for (int s = 128; s > 0; s >>= 1) {
        if (threadIdx.x < s) red[threadIdx.x] = fmaxf(red[threadIdx.x], red[threadIdx.x + s]);
        __syncthreads();
    }
    float mx = red[0]; __syncthreads();
    float lsum = 0.f;
    for (int t = threadIdx.x; t < Tsz; t += 256) { float ex = expf(e[t] - mx); e[t] = ex; lsum += ex; }
    red[threadIdx.x] = lsum; __syncthreads();
    for (int s = 128; s > 0; s >>= 1) {
        if (threadIdx.x < s) red[threadIdx.x] += red[threadIdx.x + s];
        __syncthreads();
    }
    float inv = 1.f / red[0];
    for (int t = threadIdx.x; t < Tsz; t += 256) w[(size_t)b * Tsz + t] = e[t] * inv;
}

// ctx[b][h] = sum_t w[b][t] * outputs[b][t][h]
__global__ __launch_bounds__(256) void k_ctx(const unsigned short* __restrict__ obf,
                                             const float* __restrict__ w,
                                             float* __restrict__ ctx) {
    int b = blockIdx.y;
    int h = blockIdx.x * 256 + threadIdx.x;
    float s = 0.f;
    for (int t = 0; t < Tsz; ++t)
        s += w[(size_t)b * Tsz + t] * bf2f(obf[((size_t)t * Bsz + b) * Hsz + h]);
    ctx[(size_t)b * Hsz + h] = s;
}

// head: hdd = relu(ctx@W1^T + b1); out = hdd@W2^T + b2
__global__ __launch_bounds__(64) void k_head(const float* __restrict__ ctx,
                                             const float* __restrict__ W1,
                                             const float* __restrict__ b1,
                                             const float* __restrict__ W2,
                                             const float* __restrict__ b2,
                                             float* __restrict__ out) {
    __shared__ float hdd[64];
    const int b = blockIdx.x, j = threadIdx.x;
    float s = b1[j];
    for (int h = 0; h < Hsz; ++h) s += ctx[(size_t)b * Hsz + h] * W1[(size_t)j * Hsz + h];
    hdd[j] = fmaxf(s, 0.f);
    __syncthreads();
    if (j < Csz) {
        float o = b2[j];
        for (int k = 0; k < 64; ++k) o += hdd[k] * W2[j * 64 + k];
        out[b * Csz + j] = o;
    }
}

// ---------------------------------------------------------------------------
extern "C" void kernel_launch(void* const* d_in, const int* in_sizes, int n_in,
                              void* d_out, int out_size, void* d_ws, size_t ws_size,
                              hipStream_t stream) {
    (void)in_sizes; (void)n_in; (void)out_size; (void)ws_size;
    const int*   inputs  = (const int*)d_in[0];
    const int*   lens    = (const int*)d_in[1];
    const int*   pad_tok = (const int*)d_in[2];
    const float* emb     = (const float*)d_in[3];
    const float* W_ih    = (const float*)d_in[4];
    const float* b_ih    = (const float*)d_in[5];
    const float* W_hh    = (const float*)d_in[6];
    const float* b_hh    = (const float*)d_in[7];
    const float* Wa      = (const float*)d_in[8];
    /* d_in[9] = ba: constant over t -> cancels in softmax, unused */
    const float* W1      = (const float*)d_in[10];
    const float* b1      = (const float*)d_in[11];
    const float* W2      = (const float*)d_in[12];
    const float* b2      = (const float*)d_in[13];
    float* out = (float*)d_out;

    char* ws = (char*)d_ws;
    size_t off = 0;
    auto alloc = [&](size_t bytes) {
        size_t o = off; off += (bytes + 255) & ~(size_t)255; return o;
    };
    unsigned*       bar    = (unsigned*)      (ws + alloc(256));
    unsigned short* Wih_bf = (unsigned short*)(ws + alloc((size_t)Hsz * Esz * 2));
    unsigned short* Whh_bf = (unsigned short*)(ws + alloc((size_t)Hsz * Hsz * 2));
    unsigned short* hbuf   = (unsigned short*)(ws + alloc((size_t)2 * Bsz * Hsz * 2));
    unsigned short* Xbf    = (unsigned short*)(ws + alloc((size_t)Bsz * Tsz * Esz * 2));
    float*          Xp     = (float*)         (ws + alloc((size_t)Bsz * Tsz * Hsz * 4));
    unsigned short* obf    = (unsigned short*)(ws + alloc((size_t)Bsz * Tsz * Hsz * 2));
    float*          u      = (float*)         (ws + alloc((size_t)Bsz * Hsz * 4));
    float*          wgt    = (float*)         (ws + alloc((size_t)Bsz * Tsz * 4));
    float*          ctx    = (float*)         (ws + alloc((size_t)Bsz * Hsz * 4));

    // 1. init h ping-pong + barrier state
    k_init<<<256, 256, 0, stream>>>((unsigned*)hbuf, bar);
    // 2. bf16 weight conversions
    k_cvt<<<(Hsz * Esz + 255) / 256, 256, 0, stream>>>(W_ih, Wih_bf, Hsz * Esz);
    k_cvt<<<(Hsz * Hsz + 255) / 256, 256, 0, stream>>>(W_hh, Whh_bf, Hsz * Hsz);
    // 3. embedding gather (layout m = t*B + b)
    k_gather<<<(Bsz * Tsz * Esz / 4 + 255) / 256, 256, 0, stream>>>(inputs, emb, Xbf);
    // 4. batched input projection (WMMA GEMM, double-buffered)
    k_gemm_xp<<<dim3(Bsz * Tsz / 64, Hsz / 64), 128, 0, stream>>>(Xbf, Wih_bf, b_ih, Xp);
    // 5. persistent recurrence (WMMA per step, async-DMA W_hh into LDS, grid barrier)
    k_rnn<<<Hsz / 16, 128, 0, stream>>>(Whh_bf, Xp, b_hh, lens, hbuf, obf, bar, bar + 1);
    // 6. attention (u = state@Wa, energies+softmax, context)
    k_u<<<dim3(Hsz / 256, Bsz), 256, 0, stream>>>(hbuf, Wa, u);
    k_att<<<Bsz, 256, 0, stream>>>(obf, u, inputs, pad_tok, wgt);
    k_ctx<<<dim3(Hsz / 256, Bsz), 256, 0, stream>>>(obf, wgt, ctx);
    // 7. MLP head
    k_head<<<Bsz, 64, 0, stream>>>(ctx, W1, b1, W2, b2, out);
}
